// DecoderLSTM_13967233646807
// MI455X (gfx1250) — matrix operand order repaired and checked by reading, not compiled, not run on hardware
//
#include <hip/hip_runtime.h>
#include <math.h>

// ---------------------------------------------------------------------------
// DecoderLSTM (Show-Attend-Tell) for MI455X / gfx1250, wave32 + WMMA f16.
//
// Precision strategy: inputs are fp32; weights/activations of the large GEMMs
// are converted once to f16 (A row-major, B pre-swizzled into the WMMA
// 16x16x32 B-register layout) and accumulated in f32 via
// v_wmma_f32_16x16x32_f16.  Softmax / LSTM nonlinearities stay fp32 VALU.
//
// This revision adds N-register-blocking (4 tiles/wave; 5 for the vocab GEMM)
// so one A fragment feeds 4-5 WMMAs: VMEM-per-WMMA drops from 4 to ~2.5.
//
// Workspace budget (~107 MB):
//   enc_h (sorted f16)        51.4 MB
//   Wa_enc (f32)              25.7 MB
//   packed W_ih|W_hh (f16)    12.6 MB
//   packed fc_W (f16)         10.2 MB
//   packed W_a / U_a, emb_h, state buffers, etc.   ~7 MB
// ---------------------------------------------------------------------------

typedef __attribute__((ext_vector_type(16))) _Float16 v16h;
typedef __attribute__((ext_vector_type(8)))  _Float16 v8h;
typedef __attribute__((ext_vector_type(8)))  float    v8f;

#define NB 64      // batch
#define NL 196     // spatial locations
#define NT 32      // caption length
#define ED 2048    // encoder dim
#define AD 512     // attention dim
#define DD 512     // decoder dim
#define EMBD 512   // embed dim
#define NV 10000   // vocab
#define MAXT 31    // decode steps

// ---------------------------------------------------------------------------
// 0. Stable descending argsort of lengths (B=64), gather caps, emit outputs
// ---------------------------------------------------------------------------
__global__ void sort_kernel(const int* __restrict__ lengths,
                            const int* __restrict__ captions,
                            int* __restrict__ s_idx, int* __restrict__ dl_i,
                            int* __restrict__ caps_s,
                            float* __restrict__ out_caps,
                            float* __restrict__ out_dl,
                            float* __restrict__ out_sidx) {
  __shared__ int sRank[NB];
  int i = threadIdx.x;
  if (i < NB) {
    int li = lengths[i];
    int r = 0;
    for (int j = 0; j < NB; ++j) {
      int lj = lengths[j];
      if (lj > li || (lj == li && j < i)) r++;
    }
    s_idx[r] = i;
    dl_i[r] = li - 1;
    out_dl[r] = (float)(li - 1);
    out_sidx[r] = (float)i;
    sRank[i] = r;
  }
  __syncthreads();
  if (i < NB) {
    int r = sRank[i];
    for (int t = 0; t < NT; ++t) {
      int cv = captions[i * NT + t];
      caps_s[r * NT + t] = cv;
      out_caps[r * NT + t] = (float)cv;
    }
  }
}

// ---------------------------------------------------------------------------
// 1. Gather+convert encoder features to f16 (batch-sorted)
// ---------------------------------------------------------------------------
__global__ void enc_convert(const float* __restrict__ enc,
                            const int* __restrict__ s_idx,
                            _Float16* __restrict__ enc_h) {
  size_t idx = (size_t)blockIdx.x * 256 + threadIdx.x;
  const size_t per_b = (size_t)NL * ED;
  if (idx >= (size_t)NB * per_b) return;
  int b = (int)(idx / per_b);
  size_t rem = idx - (size_t)b * per_b;
  enc_h[idx] = (_Float16)enc[(size_t)s_idx[b] * per_b + rem];
}

// ---------------------------------------------------------------------------
// 2. Gather token embeddings to f16 (sorted captions)
// ---------------------------------------------------------------------------
__global__ void emb_gather(const float* __restrict__ table,
                           const int* __restrict__ caps_s,
                           _Float16* __restrict__ emb_h) {
  int idx = blockIdx.x * 256 + threadIdx.x;   // NB*NT*EMBD
  int e = idx & (EMBD - 1);
  int bt = idx >> 9;
  emb_h[idx] = (_Float16)table[(size_t)caps_s[bt] * EMBD + e];
}

// ---------------------------------------------------------------------------
// 3. Pack an fp32 weight into the WMMA 16x16x32 f16 B-register layout.
//    Logical B is Kb x Nb.  trans=0: src[k*Nb+n]; trans=1: src[n*Kb+k].
//    Tile (nt,kt) -> dst[(nt*KtTotal + ktOff + kt)*512 + lane*16 + j]
//    lane<16 : n = nt*16+lane,    elements j -> K = kt*32 + {0..7, 16..23}
//    lane>=16: n = nt*16+lane-16, elements j -> K = kt*32 + {8..15, 24..31}
// ---------------------------------------------------------------------------
__global__ void pack_b_kernel(const float* __restrict__ src,
                              _Float16* __restrict__ dst,
                              int Kb, int Nb, int trans, int ktOff, int KtTotal) {
  int gid = blockIdx.x * 256 + threadIdx.x;
  int lane = gid & 31;
  int tile = gid >> 5;
  int Kt = Kb >> 5;
  int Nt = Nb >> 4;
  if (tile >= Kt * Nt) return;
  int kt = tile % Kt;
  int nt = tile / Kt;
  int n = nt * 16 + (lane & 15);
  int k0 = kt * 32 + ((lane >> 4) << 3);
  _Float16* out = dst + ((size_t)(nt * KtTotal + ktOff + kt)) * 512 + lane * 16;
  for (int j = 0; j < 16; ++j) {
    int k = k0 + j + ((j >> 3) << 3);
    float v = trans ? src[(size_t)n * Kb + k] : src[(size_t)k * Nb + n];
    out[j] = (_Float16)v;
  }
}

// ---------------------------------------------------------------------------
// 4. mean over L (f16 in, f32 out)
// ---------------------------------------------------------------------------
__global__ void mean_kernel(const _Float16* __restrict__ enc_h,
                            float* __restrict__ mean) {
  int idx = blockIdx.x * 256 + threadIdx.x;   // NB*ED
  int b = idx >> 11, e = idx & (ED - 1);
  const _Float16* p = enc_h + (size_t)b * NL * ED + e;
  float s = 0.f;
  for (int l = 0; l < NL; ++l) s += (float)p[(size_t)l * ED];
  mean[idx] = s * (1.f / (float)NL);
}

// ---------------------------------------------------------------------------
// 5. h0 = tanh(mean@ih_W + ih_b); c0 = tanh(mean@ic_W + ic_b)  (one-time)
// ---------------------------------------------------------------------------
__global__ void init_state(const float* __restrict__ mean,
                           const float* __restrict__ ihW, const float* __restrict__ ihb,
                           const float* __restrict__ icW, const float* __restrict__ icb,
                           float* __restrict__ h, float* __restrict__ c,
                           _Float16* __restrict__ h_h) {
  int idx = blockIdx.x * 256 + threadIdx.x;   // NB*DD
  int b = idx >> 9, n = idx & (DD - 1);
  const float* mb = mean + (size_t)b * ED;
  float sh = 0.f, sc = 0.f;
  for (int k = 0; k < ED; ++k) {
    float m = mb[k];
    sh += m * ihW[(size_t)k * DD + n];
    sc += m * icW[(size_t)k * DD + n];
  }
  float hv = tanhf(sh + ihb[n]);
  float cv = tanhf(sc + icb[n]);
  h[idx] = hv; c[idx] = cv; h_h[idx] = (_Float16)hv;
}

// ---------------------------------------------------------------------------
// WMMA A-fragment load: row-major f16, per-lane pre-offset pointer.
// ---------------------------------------------------------------------------
__device__ __forceinline__ v16h load_a_frag(const _Float16* __restrict__ Ap) {
  v8h lo = *(const v8h*)(Ap);
  v8h hi = *(const v8h*)(Ap + 16);
  return __builtin_shufflevector(lo, hi, 0, 1, 2, 3, 4, 5, 6, 7,
                                 8, 9, 10, 11, 12, 13, 14, 15);
}

// ---------------------------------------------------------------------------
// 6. N-blocked WMMA GEMM: C(MxN,f32) = A(MxK,f16 row-major) x Bpacked(KxN).
//    Each wave computes NBLK adjacent 16x16 N-tiles of one M-tile, so one A
//    fragment feeds NBLK WMMAs per k-step.  Nt must be a multiple of NBLK.
// ---------------------------------------------------------------------------
template <int NBLK>
__global__ void wmma_gemm(const _Float16* __restrict__ A,
                          const _Float16* __restrict__ Bp,
                          float* __restrict__ C, int M, int N, int K) {
  int wid = (blockIdx.x * blockDim.x + threadIdx.x) >> 5;
  int lane = threadIdx.x & 31;
  int nWaves = (gridDim.x * blockDim.x) >> 5;
  int Mt = M >> 4, Nt = N >> 4, Kt = K >> 5;
  int NtG = Nt / NBLK;
  int groups = Mt * NtG;
  for (int g = wid; g < groups; g += nWaves) {
    int mt = g % Mt;
    int ntBase = (g / Mt) * NBLK;
    const _Float16* Ap = A + (size_t)(mt * 16 + (lane & 15)) * K + ((lane >> 4) << 3);
    const _Float16* Bt[NBLK];
    for (int u = 0; u < NBLK; ++u)
      Bt[u] = Bp + (size_t)(ntBase + u) * Kt * 512 + lane * 16;
    v8f acc[NBLK];
    for (int u = 0; u < NBLK; ++u) acc[u] = (v8f){};
    for (int kt = 0; kt < Kt; ++kt) {
      v16h a = load_a_frag(Ap + (size_t)kt * 32);
      // stream packed B ahead (global_prefetch_b8; speculative, OOB-safe)
      __builtin_prefetch(Bt[0] + (size_t)(kt + 8) * 512, 0, 1);
      for (int u = 0; u < NBLK; ++u) {
        v16h bm = *(const v16h*)(Bt[u] + (size_t)kt * 512);
        acc[u] = __builtin_amdgcn_wmma_f32_16x16x32_f16(false, a, false, bm,
                                                        (short)0, acc[u], false, false);
      }
    }
    int mb = mt * 16 + ((lane >> 4) << 3);
    for (int u = 0; u < NBLK; ++u) {
      int n = (ntBase + u) * 16 + (lane & 15);
      for (int r = 0; r < 8; ++r)
        C[(size_t)(mb + r) * N + n] = acc[u][r];
    }
  }
}

// ---------------------------------------------------------------------------
// 7. Vocab projection, N-blocked by 5 (625 N-tiles = 5*125), bias + mask
//    epilogue, writes predictions[b][t][:].
// ---------------------------------------------------------------------------
__global__ void wmma_pred_gemm(const _Float16* __restrict__ A,       // hnew_h 64x512
                               const _Float16* __restrict__ Bp,      // fc packed
                               const float* __restrict__ bias,
                               const int* __restrict__ dl, int t,
                               float* __restrict__ out) {            // preds base
  const int M = NB, N = NV, K = DD;
  const int Mt = M >> 4, Nt = N >> 4, Kt = K >> 5;
  const int NBLK = 5;
  const int NtG = Nt / NBLK;                  // 125
  int wid = (blockIdx.x * blockDim.x + threadIdx.x) >> 5;
  int lane = threadIdx.x & 31;
  int nWaves = (gridDim.x * blockDim.x) >> 5;
  int groups = Mt * NtG;                      // 500
  for (int g = wid; g < groups; g += nWaves) {
    int mt = g % Mt;
    int ntBase = (g / Mt) * NBLK;
    const _Float16* Ap = A + (size_t)(mt * 16 + (lane & 15)) * K + ((lane >> 4) << 3);
    const _Float16* Bt[NBLK];
    for (int u = 0; u < NBLK; ++u)
      Bt[u] = Bp + (size_t)(ntBase + u) * Kt * 512 + lane * 16;
    v8f acc[NBLK];
    for (int u = 0; u < NBLK; ++u) acc[u] = (v8f){};
    for (int kt = 0; kt < Kt; ++kt) {
      v16h a = load_a_frag(Ap + (size_t)kt * 32);
      __builtin_prefetch(Bt[0] + (size_t)(kt + 8) * 512, 0, 1);
      for (int u = 0; u < NBLK; ++u) {
        v16h bm = *(const v16h*)(Bt[u] + (size_t)kt * 512);
        acc[u] = __builtin_amdgcn_wmma_f32_16x16x32_f16(false, a, false, bm,
                                                        (short)0, acc[u], false, false);
      }
    }
    int mb = mt * 16 + ((lane >> 4) << 3);
    for (int u = 0; u < NBLK; ++u) {
      int n = (ntBase + u) * 16 + (lane & 15);
      float bn = bias[n];
      for (int r = 0; r < 8; ++r) {
        int b = mb + r;
        float v = (t < dl[b]) ? (acc[u][r] + bn) : 0.f;
        out[((size_t)b * MAXT + t) * NV + n] = v;
      }
    }
  }
}

// ---------------------------------------------------------------------------
// 8. Attention energy: one wave per (b,l): dot(tanh(WaEnc+hU), w_att)
// ---------------------------------------------------------------------------
__global__ void attn_energy(const float* __restrict__ WaEnc,
                            const float* __restrict__ hU,
                            const float* __restrict__ w_att,
                            float* __restrict__ energy) {
  int wid = (blockIdx.x * blockDim.x + threadIdx.x) >> 5;
  int lane = threadIdx.x & 31;
  if (wid >= NB * NL) return;
  int b = wid / NL;
  const float* row = WaEnc + (size_t)wid * AD;
  const float* hu = hU + (size_t)b * AD;
  float s = 0.f;
  for (int a = lane; a < AD; a += 32)
    s += tanhf(row[a] + hu[a]) * w_att[a];
  for (int off = 16; off; off >>= 1) s += __shfl_xor(s, off, 32);
  if (lane == 0) energy[wid] = s;
}

// ---------------------------------------------------------------------------
// 9. Per-batch softmax + gated context + assemble x = [emb_t | b*ctx | h] f16
// ---------------------------------------------------------------------------
__global__ void softmax_context(const float* __restrict__ energy,
                                const float* __restrict__ h,
                                const float* __restrict__ fbW,
                                const float* __restrict__ fbb,
                                const _Float16* __restrict__ enc_h,
                                const _Float16* __restrict__ emb_h,
                                const _Float16* __restrict__ h_h,
                                const int* __restrict__ dl, int t,
                                float* __restrict__ alphas_out,
                                _Float16* __restrict__ x_h) {
  int b = blockIdx.x;
  int tid = threadIdx.x;                       // 256 threads
  __shared__ float sE[NL];
  __shared__ float sRed[256];
  __shared__ float sBeta, sMax, sSum;

  // beta = sigmoid(h . fbW + fbb)
  float p = 0.f;
  for (int i = tid; i < DD; i += 256) p += h[(size_t)b * DD + i] * fbW[i];
  sRed[tid] = p; __syncthreads();
  for (int s = 128; s; s >>= 1) { if (tid < s) sRed[tid] += sRed[tid + s]; __syncthreads(); }
  if (tid == 0) sBeta = 1.f / (1.f + expf(-(sRed[0] + fbb[0])));
  __syncthreads();

  for (int i = tid; i < NL; i += 256) sE[i] = energy[(size_t)b * NL + i];
  __syncthreads();

  float m = -1e30f;
  for (int i = tid; i < NL; i += 256) m = fmaxf(m, sE[i]);
  sRed[tid] = m; __syncthreads();
  for (int s = 128; s; s >>= 1) { if (tid < s) sRed[tid] = fmaxf(sRed[tid], sRed[tid + s]); __syncthreads(); }
  if (tid == 0) sMax = sRed[0];
  __syncthreads();

  float es = 0.f;
  for (int i = tid; i < NL; i += 256) { float v = expf(sE[i] - sMax); sE[i] = v; es += v; }
  sRed[tid] = es; __syncthreads();
  for (int s = 128; s; s >>= 1) { if (tid < s) sRed[tid] += sRed[tid + s]; __syncthreads(); }
  if (tid == 0) sSum = sRed[0];
  __syncthreads();

  int mask = (t < dl[b]);
  float inv = 1.f / sSum;
  for (int i = tid; i < NL; i += 256) {
    float a = sE[i] * inv;
    sE[i] = a;
    alphas_out[((size_t)b * MAXT + t) * NL + i] = mask ? a : 0.f;
  }
  __syncthreads();

  // context[e] = beta * sum_l alpha[l] * enc_h[b][l][e]; e = tid + j*256
  float acc[ED / 256] = {};
  const _Float16* eb = enc_h + (size_t)b * NL * ED;
  for (int l = 0; l < NL; ++l) {
    float a = sE[l];
    const _Float16* row = eb + (size_t)l * ED;
    for (int j = 0; j < ED / 256; ++j)
      acc[j] += a * (float)row[tid + j * 256];
  }
  _Float16* xb = x_h + (size_t)b * (EMBD + ED + DD);   // 3072
  float beta = sBeta;
  for (int j = 0; j < ED / 256; ++j)
    xb[EMBD + tid + j * 256] = (_Float16)(beta * acc[j]);
  for (int i = tid; i < EMBD; i += 256) {
    xb[i] = emb_h[((size_t)b * NT + t) * EMBD + i];
    xb[EMBD + ED + i] = h_h[(size_t)b * DD + i];
  }
}

// ---------------------------------------------------------------------------
// 10. LSTM cell elementwise + masked state update; refresh f16 copies
// ---------------------------------------------------------------------------
__global__ void lstm_cell(const float* __restrict__ gates,
                          const float* __restrict__ b_ih,
                          const float* __restrict__ b_hh,
                          float* __restrict__ h, float* __restrict__ c,
                          _Float16* __restrict__ h_h,
                          _Float16* __restrict__ hnew_h,
                          const int* __restrict__ dl, int t) {
  int idx = blockIdx.x * 256 + threadIdx.x;   // NB*DD
  int b = idx >> 9, j = idx & (DD - 1);
  const float* g = gates + (size_t)b * (4 * DD);
  float gi = g[j]            + b_ih[j]            + b_hh[j];
  float gf = g[DD + j]       + b_ih[DD + j]       + b_hh[DD + j];
  float gg = g[2 * DD + j]   + b_ih[2 * DD + j]   + b_hh[2 * DD + j];
  float go = g[3 * DD + j]   + b_ih[3 * DD + j]   + b_hh[3 * DD + j];
  float iv = 1.f / (1.f + expf(-gi));
  float fv = 1.f / (1.f + expf(-gf));
  float gv = tanhf(gg);
  float ov = 1.f / (1.f + expf(-go));
  float cn = fv * c[idx] + iv * gv;
  float hn = ov * tanhf(cn);
  bool m = (t < dl[b]);
  float h2 = m ? hn : h[idx];
  float c2 = m ? cn : c[idx];
  h[idx] = h2; c[idx] = c2;
  h_h[idx] = (_Float16)h2;
  hnew_h[idx] = (_Float16)hn;
}

// ---------------------------------------------------------------------------
// host orchestration
// ---------------------------------------------------------------------------
extern "C" void kernel_launch(void* const* d_in, const int* in_sizes, int n_in,
                              void* d_out, int out_size, void* d_ws, size_t ws_size,
                              hipStream_t stream) {
  const float* enc      = (const float*)d_in[0];
  const int*   captions = (const int*)d_in[1];
  const int*   lengths  = (const int*)d_in[2];
  const float* emb_tab  = (const float*)d_in[3];
  const float* W_a      = (const float*)d_in[4];
  const float* U_a      = (const float*)d_in[5];
  const float* w_att    = (const float*)d_in[6];
  const float* fb_W     = (const float*)d_in[7];
  const float* fb_b     = (const float*)d_in[8];
  const float* W_ih     = (const float*)d_in[9];
  const float* W_hh     = (const float*)d_in[10];
  const float* b_ih     = (const float*)d_in[11];
  const float* b_hh     = (const float*)d_in[12];
  const float* fc_W     = (const float*)d_in[13];
  const float* fc_b     = (const float*)d_in[14];
  const float* ih_W     = (const float*)d_in[15];
  const float* ih_b     = (const float*)d_in[16];
  const float* ic_W     = (const float*)d_in[17];
  const float* ic_b     = (const float*)d_in[18];
  (void)in_sizes; (void)n_in; (void)out_size; (void)ws_size;

  // output layout: predictions | caps | dl | alphas | sort_idx (all as f32)
  float* out       = (float*)d_out;
  float* out_pred  = out;
  float* out_caps  = out_pred + (size_t)NB * MAXT * NV;
  float* out_dl    = out_caps + (size_t)NB * NT;
  float* out_alph  = out_dl + NB;
  float* out_sidx  = out_alph + (size_t)NB * MAXT * NL;

  // workspace carve-up
  char* w = (char*)d_ws;
  auto alloc = [&](size_t bytes) -> void* {
    void* p = (void*)w;
    w += (bytes + 255) & ~(size_t)255;
    return p;
  };
  int*      s_idx   = (int*)alloc(NB * 4);
  int*      dl_i    = (int*)alloc(NB * 4);
  int*      caps_s  = (int*)alloc((size_t)NB * NT * 4);
  _Float16* enc_h   = (_Float16*)alloc((size_t)NB * NL * ED * 2);
  _Float16* Wa_p    = (_Float16*)alloc((size_t)(AD / 16) * (ED / 32) * 512 * 2);
  _Float16* Ua_p    = (_Float16*)alloc((size_t)(AD / 16) * (DD / 32) * 512 * 2);
  _Float16* Wc_p    = (_Float16*)alloc((size_t)(4 * DD / 16) * (3072 / 32) * 512 * 2);
  _Float16* fc_p    = (_Float16*)alloc((size_t)(NV / 16) * (DD / 32) * 512 * 2);
  _Float16* emb_h   = (_Float16*)alloc((size_t)NB * NT * EMBD * 2);
  float*    WaEnc   = (float*)alloc((size_t)NB * NL * AD * 4);
  float*    mean    = (float*)alloc((size_t)NB * ED * 4);
  float*    hbuf    = (float*)alloc((size_t)NB * DD * 4);
  float*    cbuf    = (float*)alloc((size_t)NB * DD * 4);
  float*    hU      = (float*)alloc((size_t)NB * AD * 4);
  _Float16* h_h     = (_Float16*)alloc((size_t)NB * DD * 2);
  _Float16* hnew_h  = (_Float16*)alloc((size_t)NB * DD * 2);
  _Float16* x_h     = (_Float16*)alloc((size_t)NB * 3072 * 2);
  float*    energy  = (float*)alloc((size_t)NB * NL * 4);
  float*    gates   = (float*)alloc((size_t)NB * 4 * DD * 4);

  // --- setup phase (one-time) ---
  sort_kernel<<<1, 64, 0, stream>>>(lengths, captions, s_idx, dl_i, caps_s,
                                    out_caps, out_dl, out_sidx);

  {
    size_t n = (size_t)NB * NL * ED;
    enc_convert<<<(int)((n + 255) / 256), 256, 0, stream>>>(enc, s_idx, enc_h);
  }
  emb_gather<<<(NB * NT * EMBD) / 256, 256, 0, stream>>>(emb_tab, caps_s, emb_h);

  // pack weights into WMMA-B layout (f16)
  // W_a: B = 2048x512, K-major
  pack_b_kernel<<<(32 * (ED / 32) * (AD / 16)) / 256 + 1, 256, 0, stream>>>(
      W_a, Wa_p, ED, AD, 0, 0, ED / 32);
  // U_a: B = 512x512, K-major
  pack_b_kernel<<<(32 * (DD / 32) * (AD / 16)) / 256 + 1, 256, 0, stream>>>(
      U_a, Ua_p, DD, AD, 0, 0, DD / 32);
  // W_ih^T: B = 2560x2048 (src row-major 2048x2560) -> ktiles [0,80)
  pack_b_kernel<<<(32 * (2560 / 32) * (4 * DD / 16)) / 256 + 1, 256, 0, stream>>>(
      W_ih, Wc_p, 2560, 4 * DD, 1, 0, 3072 / 32);
  // W_hh^T: B = 512x2048 -> ktiles [80,96)
  pack_b_kernel<<<(32 * (DD / 32) * (4 * DD / 16)) / 256 + 1, 256, 0, stream>>>(
      W_hh, Wc_p, DD, 4 * DD, 1, 2560 / 32, 3072 / 32);
  // fc_W: B = 512x10000, K-major
  pack_b_kernel<<<(32 * (DD / 32) * (NV / 16)) / 256 + 1, 256, 0, stream>>>(
      fc_W, fc_p, DD, NV, 0, 0, DD / 32);

  mean_kernel<<<(NB * ED) / 256, 256, 0, stream>>>(enc_h, mean);
  init_state<<<(NB * DD) / 256, 256, 0, stream>>>(mean, ih_W, ih_b, ic_W, ic_b,
                                                  hbuf, cbuf, h_h);

  // Wa_enc = enc_h @ W_a : M=12544, N=512, K=2048
  // groups = 784 Mt * (32/4) NtG = 6272 waves -> 784 blocks
  wmma_gemm<4><<<784, 256, 0, stream>>>(enc_h, Wa_p, WaEnc, NB * NL, AD, ED);

  // --- time loop: 31 steps ---
  for (int t = 0; t < MAXT; ++t) {
    // hU = h @ U_a : 64x512x512 -> 4*8 = 32 groups -> 4 blocks
    wmma_gemm<4><<<4, 256, 0, stream>>>(h_h, Ua_p, hU, NB, AD, DD);

    // energy[b][l] = dot(tanh(WaEnc+hU), w_att) : one wave per (b,l)
    attn_energy<<<(NB * NL * 32) / 256, 256, 0, stream>>>(WaEnc, hU, w_att, energy);

    // softmax + gated context + assemble x = [emb_t | beta*ctx | h]
    softmax_context<<<NB, 256, 0, stream>>>(energy, hbuf, fb_W, fb_b, enc_h,
                                            emb_h, h_h, dl_i, t, out_alph, x_h);

    // gates = x @ [W_ih|W_hh]^T : 64x3072x2048 -> 4*32 = 128 groups -> 16 blocks
    wmma_gemm<4><<<16, 256, 0, stream>>>(x_h, Wc_p, gates, NB, 4 * DD, 3072);

    // LSTM cell + masked update
    lstm_cell<<<(NB * DD) / 256, 256, 0, stream>>>(gates, b_ih, b_hh, hbuf, cbuf,
                                                   h_h, hnew_h, dl_i, t);

    // predictions[:, t, :] = mask ? hnew @ fc_W + fc_b : 0
    // groups = 4 * 125 = 500 waves -> 63 blocks (grid-stride guarded)
    wmma_pred_gemm<<<63, 256, 0, stream>>>(hnew_h, fc_p, fc_b, dl_i, t, out_pred);
  }
}
